// PixelRnn_40003325395514
// MI455X (gfx1250) — matrix operand order repaired and checked by reading, not compiled, not run on hardware
//
#include <hip/hip_runtime.h>
#include <hip/hip_bf16.h>

// ---------------------------------------------------------------------------
// PixelRNN (4x directional GRU over rows/cols) for gfx1250 (MI455X, wave32).
//  * prep kernels convert x and all 8 weight matrices to f16.
//  * ONE fused GRU kernel covers all 4 directions (gridDim.y): each block owns
//    16 batch rows of one direction; 16 waves each own a 16-wide hidden-column
//    tile; per time step each wave issues 48 v_wmma_f32_16x16x32_f16
//    (h@Whh^T fused with xs@Wih^T), gate math in f32, h double-buffered in
//    LDS (f16), one barrier per step.
//  * Weights (768 KB/block working set) cannot be register- or LDS-resident;
//    they are streamed from L2 every step. A per-step compiler memory barrier
//    stops the compiler from hoisting the loads and spilling them to scratch
//    (which v1 did).
// ---------------------------------------------------------------------------

typedef __attribute__((ext_vector_type(16))) _Float16 v16h;
typedef __attribute__((ext_vector_type(8)))  _Float16 v8h;
typedef __attribute__((ext_vector_type(8)))  float    v8f;

namespace cfg {
constexpr int B = 8, C = 256, H = 64, W = 64, O = 256;
constexpr int T = 64;          // sequence length (W horizontal, H vertical)
constexpr int N = 512;         // batch of sequences (B*H or B*W)
constexpr int G = 3 * O;       // 768 gate rows
constexpr int LDH = 264;       // LDS row stride in halves (16B-aligned, skewed)
constexpr int WSZ = G * C;     // elements per weight matrix (196608)
}

// ---- WMMA fragment helpers (layouts per CDNA5 ISA 7.12.2) ------------------

__device__ __forceinline__ v16h pack16(v8h lo, v8h hi) {
  union { v16h v; v8h h[2]; } u;
  u.h[0] = lo; u.h[1] = hi;
  return u.v;
}

// A-matrix 16x32 f16 fragment from row-major [16 x ld] source.
// lane 0-15: M=lane,    K = {k0+0..7,  k0+16..23}
// lane16-31: M=lane-16, K = {k0+8..15, k0+24..31}
__device__ __forceinline__ v16h load_a_frag(const _Float16* base, int ld, int k0) {
  const int lane = threadIdx.x & 31;
  const int hlf  = lane >> 4;
  const int m    = lane & 15;
  const _Float16* p = base + m * ld + k0 + 8 * hlf;
  return pack16(*(const v8h*)p, *(const v8h*)(p + 16));
}

// B-matrix 32x16 f16 fragment where B[k][n] = Wt[g0+n][k0+k] (Wt row-major,
// row stride ld). lane 0-15: N=lane, K={k0+0..7, k0+16..23}; lane 16-31: N=lane-16, K={k0+8..15, k0+24..31}.
__device__ __forceinline__ v16h load_b_frag(const _Float16* Wt, int ld, int g0, int k0) {
  const int lane = threadIdx.x & 31;
  const int hlf  = lane >> 4;
  const int n    = lane & 15;
  const _Float16* p = Wt + (size_t)(g0 + n) * ld + k0 + 8 * hlf;
  return pack16(*(const v8h*)p, *(const v8h*)(p + 16));
}

__device__ __forceinline__ v8f wmma16(v16h a, v16h b, v8f c) {
  return __builtin_amdgcn_wmma_f32_16x16x32_f16(false, a, false, b, (short)0, c,
                                                false, false);
}

__device__ __forceinline__ float sigm(float x) {
  return 1.0f / (1.0f + __expf(-x));
}

// ---- prep: f32 -> f16 weight conversion ------------------------------------

__global__ __launch_bounds__(256) void cvt_weights_kernel(
    const float* __restrict__ s0, const float* __restrict__ s1,
    const float* __restrict__ s2, const float* __restrict__ s3,
    const float* __restrict__ s4, const float* __restrict__ s5,
    const float* __restrict__ s6, const float* __restrict__ s7,
    _Float16* __restrict__ dst) {
  const int total = 8 * cfg::WSZ;
  int i = blockIdx.x * blockDim.x + threadIdx.x;
  if (i >= total) return;
  const int m = i / cfg::WSZ;
  const int o = i - m * cfg::WSZ;
  const float* s;
  switch (m) {
    case 0: s = s0; break; case 1: s = s1; break;
    case 2: s = s2; break; case 3: s = s3; break;
    case 4: s = s4; break; case 5: s = s5; break;
    case 6: s = s6; break; default: s = s7; break;
  }
  dst[i] = (_Float16)s[o];
}

// ---- prep: x [B,C,H,W] f32 -> xs_h [W][B*H][C] f16 and xs_v [H][B*W][C] f16 -

__global__ __launch_bounds__(256) void build_xs_kernel(
    const float* __restrict__ x,
    _Float16* __restrict__ xs_h,
    _Float16* __restrict__ xs_v) {
  using namespace cfg;
  const int total = B * C * H * W;
  int i = blockIdx.x * blockDim.x + threadIdx.x;
  if (i >= total) return;
  const int w  = i & 63;
  const int h  = (i >> 6) & 63;
  const int c  = (i >> 12) & 255;
  const int b  = i >> 20;
  const _Float16 v = (_Float16)x[i];
  xs_h[((size_t)w * N + (b * H + h)) * C + c] = v;
  xs_v[((size_t)h * N + (b * W + w)) * C + c] = v;
}

// ---- fused GRU kernel, all 4 directions -------------------------------------
// grid = (N/16, 4); block = 512 threads = 16 waves.
// blockIdx.x: 16-row batch tile; blockIdx.y: direction (0 h_f, 1 h_b, 2 v_f, 3 v_b).
// Wave j owns hidden columns [16j, 16j+16).

struct DirArgs {
  const _Float16* xs;    // [T][N][C] f16
  const _Float16* Wih;   // [G][C] f16
  const _Float16* Whh;   // [G][C] f16
  const float*    bih;   // [G] f32
  const float*    bhh;   // [G] f32
};
struct AllArgs { DirArgs d[4]; };

__global__ __launch_bounds__(512) void gru_all_kernel(AllArgs args,
                                                      float* __restrict__ out) {
  using namespace cfg;

  __shared__ __attribute__((aligned(16))) _Float16 hbuf[2][16 * LDH];

  const int  dir  = blockIdx.y;
  const bool VERT = dir >= 2;
  const bool BACK = dir & 1;
  const DirArgs a = args.d[dir];

  const int wave = threadIdx.x >> 5;     // 0..15 -> column tile
  const int lane = threadIdx.x & 31;
  const int hlf  = lane >> 4;
  const int ncol = lane & 15;            // column within tile (C/D layout)
  const int n0   = blockIdx.x * 16;      // batch-row tile
  const int j0   = wave * 16;
  const int j    = j0 + ncol;            // global hidden column

  // Per-column fused biases.
  const float br  = a.bih[j]         + a.bhh[j];
  const float bz  = a.bih[O + j]     + a.bhh[O + j];
  const float bxn = a.bih[2 * O + j];
  const float bhn = a.bhh[2 * O + j];

  const int chan = (VERT ? 2 * O : 0) + (BACK ? O : 0) + j;

  // h0 = 0
  for (int i = threadIdx.x; i < 16 * LDH; i += blockDim.x)
    hbuf[0][i] = (_Float16)0.0f;
  __syncthreads();

  int p = 0;
  for (int s = 0; s < T; ++s) {
    // Forbid hoisting loop-"invariant" weight loads out of the time loop:
    // they cannot all live in registers (768 KB working set per block) and
    // would be spilled to scratch. Streaming them from L2 each step is the
    // intended data path.
    asm volatile("" ::: "memory");

    const int t = BACK ? (T - 1 - s) : s;
    const _Float16* xrow = a.xs + ((size_t)t * N + n0) * C;
    const _Float16* hrow = &hbuf[p][0];

    // Prefetch next step's xs tile toward L2 (global_prefetch_b8).
    if (s + 1 < T) {
      const int tn = BACK ? (t - 1) : (t + 1);
      __builtin_prefetch(a.xs + ((size_t)tn * N + n0) * C + lane * 64, 0, 1);
    }

    v8f accR = {}, accZ = {}, accXN = {}, accHN = {};
#pragma unroll
    for (int k0 = 0; k0 < C; k0 += 32) {
      const v16h ah = load_a_frag(hrow, LDH, k0);   // h tile from LDS
      const v16h ax = load_a_frag(xrow, C,   k0);   // xs tile from global
      // Group by shared A operand for back-to-back matrix issue.
      accR  = wmma16(ah, load_b_frag(a.Whh, C, j0,         k0), accR);
      accZ  = wmma16(ah, load_b_frag(a.Whh, C, O + j0,     k0), accZ);
      accHN = wmma16(ah, load_b_frag(a.Whh, C, 2 * O + j0, k0), accHN);
      accR  = wmma16(ax, load_b_frag(a.Wih, C, j0,         k0), accR);
      accZ  = wmma16(ax, load_b_frag(a.Wih, C, O + j0,     k0), accZ);
      accXN = wmma16(ax, load_b_frag(a.Wih, C, 2 * O + j0, k0), accXN);
    }

    // Gates + state update. C/D layout: acc[r] is (m = r + 8*hlf, col = ncol).
#pragma unroll
    for (int r = 0; r < 8; ++r) {
      const int m  = r + 8 * hlf;
      const float rg = sigm(accR[r] + br);
      const float zg = sigm(accZ[r] + bz);
      const float ng = tanhf(accXN[r] + bxn + rg * (accHN[r] + bhn));
      const float hp = (float)hbuf[p][m * LDH + j];
      const float hn = (1.0f - zg) * ng + zg * hp;
      hbuf[p ^ 1][m * LDH + j] = (_Float16)hn;

      // Scatter to out[b, chan, hh, ww].
      const int n   = n0 + m;
      const int b   = n >> 6;
      const int pos = n & 63;
      const int hh  = VERT ? t : pos;
      const int ww  = VERT ? pos : t;
      out[(((size_t)b * (4 * O) + chan) * H + hh) * W + ww] = hn;
    }
    __syncthreads();   // writes to hbuf[p^1] visible; reads of hbuf[p] done
    p ^= 1;
  }
}

// ---------------------------------------------------------------------------

extern "C" void kernel_launch(void* const* d_in, const int* in_sizes, int n_in,
                              void* d_out, int out_size, void* d_ws, size_t ws_size,
                              hipStream_t stream) {
  using namespace cfg;
  (void)in_sizes; (void)n_in; (void)out_size; (void)ws_size;

  const float* x = (const float*)d_in[0];
  // dir order: 0 = h_f, 1 = h_b, 2 = v_f, 3 = v_b
  const float* Wih[4] = {(const float*)d_in[1], (const float*)d_in[5],
                         (const float*)d_in[9], (const float*)d_in[13]};
  const float* Whh[4] = {(const float*)d_in[2], (const float*)d_in[6],
                         (const float*)d_in[10], (const float*)d_in[14]};
  const float* bih[4] = {(const float*)d_in[3], (const float*)d_in[7],
                         (const float*)d_in[11], (const float*)d_in[15]};
  const float* bhh[4] = {(const float*)d_in[4], (const float*)d_in[8],
                         (const float*)d_in[12], (const float*)d_in[16]};
  float* out = (float*)d_out;

  // Workspace layout (halves): [8 x WSZ weights][xs_h][xs_v]  (~35 MB total)
  _Float16* w16  = (_Float16*)d_ws;
  _Float16* xs_h = w16 + (size_t)8 * WSZ;
  _Float16* xs_v = xs_h + (size_t)T * N * C;

  // 1) convert weights (dst order: Wih dirs 0..3, then Whh dirs 0..3)
  {
    const int total = 8 * WSZ;
    cvt_weights_kernel<<<(total + 255) / 256, 256, 0, stream>>>(
        Wih[0], Wih[1], Wih[2], Wih[3], Whh[0], Whh[1], Whh[2], Whh[3], w16);
  }
  // 2) build f16 sequence tensors for both scan orientations
  {
    const int total = B * C * H * W;
    build_xs_kernel<<<(total + 255) / 256, 256, 0, stream>>>(x, xs_h, xs_v);
  }

  // 3) all four recurrent passes in one launch (blockIdx.y = direction)
  AllArgs args;
  for (int d = 0; d < 4; ++d) {
    args.d[d].xs  = (d < 2) ? xs_h : xs_v;
    args.d[d].Wih = w16 + (size_t)d * WSZ;
    args.d[d].Whh = w16 + (size_t)(4 + d) * WSZ;
    args.d[d].bih = bih[d];
    args.d[d].bhh = bhh[d];
  }
  dim3 grid(N / 16, 4), block(512);
  gru_all_kernel<<<grid, block, 0, stream>>>(args, out);
}
